// NGP_61667140436303
// MI455X (gfx1250) — matrix operand order repaired and checked by reading, not compiled, not run on hardware
//
#include <hip/hip_runtime.h>
#include <hip/hip_bf16.h>

// ---------------------------------------------------------------------------
// Fused Instant-NGP forward for gfx1250 (MI455X): hash-grid encode + geo MLP +
// SH deg4 + rgb MLP. All activations LDS-resident; all 5 GEMMs via
// v_wmma_f32_16x16x32_f16. Weights are pre-swizzled once per block into
// fragment-major LDS layout so each B-fragment load is 2x ds_load_b128.
// One wave32 = one 16-point tile; 256-thread block = 8 waves = 128 points.
// ---------------------------------------------------------------------------

typedef __attribute__((ext_vector_type(16))) _Float16 v16h;
typedef __attribute__((ext_vector_type(8)))  float    v8f;

#define NLEV   16
#define TBLSZ  (1u << 19)           // T = 524288 entries/level, 2 feats each
#define NPTS_PER_BLOCK 128
#define NFRAG  22                   // total 32x16 B-tiles across all weights

// s_l = 16 * 2^(0.4*l) - 1 ; res_l = ceil(s_l)+1 ; dense iff res^3 <= T (l<6)
__constant__ float kLevS[NLEV] = {
  15.0f,               20.112126572366307f, 26.85761802547597f,  35.758347359905117f,
  47.50293012833274f,  63.0f,               83.44850628946523f,  110.43047210190387f,
  146.03338943962046f, 193.01172051333097f, 255.0f,              336.7940251578609f,
  444.72188840761552f, 587.13355775848187f, 775.04688205332384f, 1023.0f };
__constant__ unsigned kLevRes[NLEV] = {
  16u, 22u, 28u, 37u, 49u, 64u, 85u, 112u, 148u, 195u, 256u, 338u, 446u, 589u, 777u, 1024u };

// Fragment ids (each = one 32x16 B tile, stored lane-major: [lane][16 halves]):
//  0..3  : geo_w0 (32x64),      nc=frag,            krow0=0
//  4..7  : geo_w1 (64x17->32),  nc=(frag-4)>>1,     kc=(frag-4)&1
//  8..11 : rgb_w0 (32x64),      nc=frag-8,          krow0=0
// 12..19 : rgb_w1 (64x64),      nc=(frag-12)>>1,    kc=(frag-12)&1
// 20..21 : rgb_w2 (64x3->16),   nc=0,               kc=frag-20

// ---- WMMA helpers ----------------------------------------------------------

__device__ inline v8f wmma16(v16h a, v16h b, v8f c) {
  // (neg_a, A, neg_b, B, c_mod, C, reuse_a, reuse_b)
  return __builtin_amdgcn_wmma_f32_16x16x32_f16(false, a, false, b, (short)0, c,
                                                false, false);
}

// A fragment 16x32 f16 from row-major LDS [16][ld].
// ISA layout: lanes 0-15 -> M=lane, K={0..7,16..23}; lanes 16-31 -> M=lane-16,
// K={8..15,24..31}. Two contiguous 8-half runs per row -> 2x ds_load_b128.
__device__ inline v16h ldA(const _Float16* lds, int ld) {
  int lane = (int)(threadIdx.x & 31u);
  int m    = lane & 15;
  int kb   = (lane >> 4) * 8;
  const _Float16* row = lds + m * ld;
  v16h a;
#pragma unroll
  for (int h = 0; h < 8; ++h) a[h] = row[kb + h];
#pragma unroll
  for (int h = 0; h < 8; ++h) a[8 + h] = row[16 + kb + h];
  return a;
}

// B fragment from pre-swizzled fragment-major LDS: lane's 16 halves are
// contiguous and 32B-aligned -> one v16h (2x ds_load_b128).
__device__ inline v16h ldBf(const _Float16* wfrag, int frag) {
  int lane = (int)(threadIdx.x & 31u);
  return *(const v16h*)(wfrag + (frag << 9) + (lane << 4));
}

// C/D fragment 16x16 f32 -> row-major f16 LDS [16][ld] (+ optional relu).
// ISA layout: lanes 0-15 -> N=lane, M=vgpr; lanes 16-31 -> N=lane-16, M=vgpr+8.
__device__ inline void stC(_Float16* lds, int ld, int col0, v8f c, bool relu) {
  int lane = (int)(threadIdx.x & 31u);
  int nn   = lane & 15;
  int mb   = (lane >> 4) * 8;
#pragma unroll
  for (int r = 0; r < 8; ++r) {
    float v = c[r];
    if (relu) v = fmaxf(v, 0.0f);
    lds[(mb + r) * ld + col0 + nn] = (_Float16)v;
  }
}

__device__ inline float sigmoidf(float v) { return 1.0f / (1.0f + __expf(-v)); }

// Source value for swizzled-weight entry (frag, lane, half) from global f32
// weights, with zero-padding for geo_w1 (17->32 cols) and rgb_w2 (3->16 cols).
__device__ inline float wsrc(int frag, int l, int h,
                             const float* gw0, const float* gw1,
                             const float* rw0, const float* rw1,
                             const float* rw2) {
  int lm = l & 15, lk = ((l >> 4) << 3) + ((h >> 3) << 3);
  if (frag < 4) {                       // geo_w0: 32x64
    int nn = frag * 16 + lm, k = lk + h;
    return gw0[k * 64 + nn];
  } else if (frag < 8) {                // geo_w1: 64x17 padded to 64x32
    int f = frag - 4, nc = f >> 1, kc = f & 1;
    int nn = nc * 16 + lm, k = kc * 32 + lk + h;
    return nn < 17 ? gw1[k * 17 + nn] : 0.0f;
  } else if (frag < 12) {               // rgb_w0: 32x64
    int nn = (frag - 8) * 16 + lm, k = lk + h;
    return rw0[k * 64 + nn];
  } else if (frag < 20) {               // rgb_w1: 64x64
    int f = frag - 12, nc = f >> 1, kc = f & 1;
    int nn = nc * 16 + lm, k = kc * 32 + lk + h;
    return rw1[k * 64 + nn];
  } else {                              // rgb_w2: 64x3 padded to 64x16
    int kc = frag - 20;
    int nn = lm, k = kc * 32 + lk + h;
    return nn < 3 ? rw2[k * 3 + nn] : 0.0f;
  }
}

// ---------------------------------------------------------------------------

__global__ __launch_bounds__(256) void ngp_fused(
    const float* __restrict__ xin, const float* __restrict__ din,
    const float* __restrict__ table,
    const float* __restrict__ gw0, const float* __restrict__ gw1,
    const float* __restrict__ rw0, const float* __restrict__ rw1,
    const float* __restrict__ rw2,
    float* __restrict__ out, int nPts) {

  // fragment-major f16 weight tiles (shared across block): 22 * 512 halves
  __shared__ __align__(32) _Float16 wfrag[NFRAG * 512];
  // per-wave activation scratch (16 rows x 64 cols, f16)
  __shared__ __align__(64) _Float16 bufA[8][16 * 64];
  __shared__ __align__(64) _Float16 bufB[8][16 * 64];

  const int tid  = (int)threadIdx.x;
  const int lane = tid & 31;
  const int wave = tid >> 5;
  const int m    = lane & 15;         // point row within wave tile
  const int grp  = lane >> 4;         // 0: levels 0-7 + SH ; 1: levels 8-15
  const int tileBase = (int)blockIdx.x * NPTS_PER_BLOCK + wave * 16;
  const int p   = tileBase + m;
  const int pc  = p < nPts ? p : nPts - 1;

  _Float16* A  = &bufA[wave][0];
  _Float16* Bb = &bufB[wave][0];

  // ---- stage 0: weights -> LDS, pre-swizzled into B-fragment order --------
  for (int i = tid; i < NFRAG * 512; i += 256) {
    int frag = i >> 9, l = (i >> 4) & 31, h = i & 15;
    wfrag[i] = (_Float16)wsrc(frag, l, h, gw0, gw1, rw0, rw1, rw2);
  }
  __syncthreads();

  // ---- stage 1: hash-grid encode -> feat 16x32 into A[:,0..31] ------------
  {
    float x0 = xin[pc * 3 + 0], x1 = xin[pc * 3 + 1], x2 = xin[pc * 3 + 2];
    // xn = clip((x + SCALE)/(2*SCALE)) with SCALE=0.5 -> clip(x+0.5)
    float u = fminf(fmaxf(x0 + 0.5f, 0.0f), 1.0f);
    float v = fminf(fmaxf(x1 + 0.5f, 0.0f), 1.0f);
    float w = fminf(fmaxf(x2 + 0.5f, 0.0f), 1.0f);
    const float2* tab2 = (const float2*)table;

    for (int i = 0; i < 8; ++i) {            // each lane-group: 8 levels
      int l = grp * 8 + i;
      float s = kLevS[l];
      unsigned res = kLevRes[l];
      float px = fmaf(u, s, 0.5f), py = fmaf(v, s, 0.5f), pz = fmaf(w, s, 0.5f);
      float fx = floorf(px), fy = floorf(py), fz = floorf(pz);
      float wx = px - fx, wy = py - fy, wz = pz - fz;
      unsigned gx = (unsigned)(int)fx, gy = (unsigned)(int)fy, gz = (unsigned)(int)fz;
      bool dense = (l < 6);
      float f0 = 0.0f, f1 = 0.0f;
#pragma unroll
      for (int c = 0; c < 8; ++c) {
        unsigned ci = (unsigned)(c & 1), cj = (unsigned)((c >> 1) & 1), ck = (unsigned)(c >> 2);
        unsigned cx = gx + ci, cy = gy + cj, cz = gz + ck;
        unsigned idx;
        if (dense) idx = cx + cy * res + cz * res * res;
        else       idx = ((cx * 1u) ^ (cy * 2654435761u) ^ (cz * 805459861u)) & (TBLSZ - 1u);
        float2 t = tab2[(unsigned)l * TBLSZ + idx];
        float ww = (ci ? wx : 1.0f - wx) * (cj ? wy : 1.0f - wy) * (ck ? wz : 1.0f - wz);
        f0 = fmaf(ww, t.x, f0);
        f1 = fmaf(ww, t.y, f1);
      }
      A[m * 64 + 2 * l + 0] = (_Float16)f0;
      A[m * 64 + 2 * l + 1] = (_Float16)f1;
    }
  }
  __syncthreads();

  // ---- stage 2: geo layer 1: h = relu(feat @ w0)  (16x32 @ 32x64) ---------
  {
    v16h a = ldA(A, 64);
    v8f z = {};
#pragma unroll
    for (int nc = 0; nc < 4; ++nc) {
      v16h b = ldBf(wfrag, nc);
      v8f c = wmma16(a, b, z);
      stC(Bb, 64, nc * 16, c, true);
    }
  }
  __syncthreads();

  // ---- stage 3: geo layer 2: h2 = h @ w1  (16x64 @ 64x32) + sigma ---------
  {
    v16h a0 = ldA(Bb, 64);
    v16h a1 = ldA(Bb + 32, 64);
    v8f z = {};
#pragma unroll
    for (int nc = 0; nc < 2; ++nc) {
      v16h b0 = ldBf(wfrag, 4 + nc * 2 + 0);
      v16h b1 = ldBf(wfrag, 4 + nc * 2 + 1);
      v8f c = wmma16(a0, b0, z);
      c = wmma16(a1, b1, c);
      int nn = lane & 15, mb = (lane >> 4) * 8;
#pragma unroll
      for (int r = 0; r < 8; ++r) {
        A[(mb + r) * 64 + nc * 16 + nn] = (_Float16)c[r];
        if (nc == 0 && nn == 0) {            // col 0 = sigma logits (f32!)
          int pp = tileBase + mb + r;
          if (pp < nPts) out[pp] = __expf(c[r]);   // trunc_exp fwd == exp
        }
      }
    }
  }
  __syncthreads();

  // ---- stage 4: SH deg4 + rin assembly into A[:,32..63] -------------------
  if (grp == 0) {
    float dx = din[pc * 3 + 0], dy = din[pc * 3 + 1], dz = din[pc * 3 + 2];
    float rn = rsqrtf(dx * dx + dy * dy + dz * dz);
    float x = dx * rn, y = dy * rn, z = dz * rn;
    float xx = x * x, yy = y * y, zz = z * z;
    float xy = x * y, yz = y * z, xz = x * z;
    _Float16* r = A + m * 64 + 32;
    r[0]  = (_Float16)(0.28209479177387814f);
    r[1]  = (_Float16)(-0.48860251190291987f * y);
    r[2]  = (_Float16)(0.48860251190291987f * z);
    r[3]  = (_Float16)(-0.48860251190291987f * x);
    r[4]  = (_Float16)(1.0925484305920792f * xy);
    r[5]  = (_Float16)(-1.0925484305920792f * yz);
    r[6]  = (_Float16)(0.94617469575756f * zz - 0.31539156525252005f);
    r[7]  = (_Float16)(-1.0925484305920792f * xz);
    r[8]  = (_Float16)(0.5462742152960396f * (xx - yy));
    r[9]  = (_Float16)(-0.5900435899266435f * y * (3.0f * xx - yy));
    r[10] = (_Float16)(2.890611442640554f * xy * z);
    r[11] = (_Float16)(-0.4570457994644657f * y * (4.0f * zz - xx - yy));
    r[12] = (_Float16)(0.37317633259011546f * z * (2.0f * zz - 3.0f * xx - 3.0f * yy));
    r[13] = (_Float16)(-0.4570457994644657f * x * (4.0f * zz - xx - yy));
    r[14] = (_Float16)(1.445305721320277f * z * (xx - yy));
    r[15] = (_Float16)(-0.5900435899266435f * x * (xx - 3.0f * yy));
  } else {
    // rin[:,16..31] = h2[:,1..16]
#pragma unroll
    for (int j = 0; j < 16; ++j) A[m * 64 + 48 + j] = A[m * 64 + 1 + j];
  }
  __syncthreads();

  // ---- stage 5: rgb layer 0: t0 = relu(rin @ rw0)  (16x32 @ 32x64) --------
  {
    v16h a = ldA(A + 32, 64);
    v8f z = {};
#pragma unroll
    for (int nc = 0; nc < 4; ++nc) {
      v16h b = ldBf(wfrag, 8 + nc);
      v8f c = wmma16(a, b, z);
      stC(Bb, 64, nc * 16, c, true);
    }
  }
  __syncthreads();

  // ---- stage 6: rgb layer 1: t1 = relu(t0 @ rw1)  (16x64 @ 64x64) ---------
  {
    v16h a0 = ldA(Bb, 64);
    v16h a1 = ldA(Bb + 32, 64);
    v8f z = {};
#pragma unroll
    for (int nc = 0; nc < 4; ++nc) {
      v16h b0 = ldBf(wfrag, 12 + nc * 2 + 0);
      v16h b1 = ldBf(wfrag, 12 + nc * 2 + 1);
      v8f c = wmma16(a0, b0, z);
      c = wmma16(a1, b1, c);
      stC(A, 64, nc * 16, c, true);
    }
  }
  __syncthreads();

  // ---- stage 7: rgb layer 2: sigmoid(t1 @ rw2) -> rgbs  (16x64 @ 64x16) ---
  {
    v16h a0 = ldA(A, 64);
    v16h a1 = ldA(A + 32, 64);
    v8f z = {};
    v16h b0 = ldBf(wfrag, 20);
    v16h b1 = ldBf(wfrag, 21);
    v8f c = wmma16(a0, b0, z);
    c = wmma16(a1, b1, c);
    int nn = lane & 15, mb = (lane >> 4) * 8;
    if (nn < 3) {
#pragma unroll
      for (int r = 0; r < 8; ++r) {
        int pp = tileBase + mb + r;
        if (pp < nPts) out[nPts + pp * 3 + nn] = sigmoidf(c[r]);
      }
    }
  }
}

// ---------------------------------------------------------------------------

extern "C" void kernel_launch(void* const* d_in, const int* in_sizes, int n_in,
                              void* d_out, int out_size, void* d_ws, size_t ws_size,
                              hipStream_t stream) {
  const float* x    = (const float*)d_in[0];
  const float* d    = (const float*)d_in[1];
  const float* tab  = (const float*)d_in[2];
  const float* gw0  = (const float*)d_in[3];
  const float* gw1  = (const float*)d_in[4];
  const float* rw0  = (const float*)d_in[5];
  const float* rw1  = (const float*)d_in[6];
  const float* rw2  = (const float*)d_in[7];
  float* out = (float*)d_out;

  int nPts = in_sizes[0] / 3;
  int blocks = (nPts + NPTS_PER_BLOCK - 1) / NPTS_PER_BLOCK;
  ngp_fused<<<blocks, 256, 0, stream>>>(x, d, tab, gw0, gw1, rw0, rw1, rw2, out, nPts);
}
